// RayTracing_9758165696555
// MI455X (gfx1250) — matrix operand order repaired and checked by reading, not compiled, not run on hardware
//
#include <hip/hip_runtime.h>
#include <hip/hip_fp16.h>
#include <math.h>

typedef __attribute__((ext_vector_type(16))) _Float16 v16h;
typedef __attribute__((ext_vector_type(8)))  _Float16 v8h;
typedef __attribute__((ext_vector_type(8)))  float    v8f;
typedef __attribute__((ext_vector_type(4)))  float    v4f;

#define HDIM 128
#define NSTEPS 64
#define SPHERE_ITERS 10
#define NSECANT 8
#define SDF_THR 5e-5f
#define EPS_F 1e-10f
#define WAVES_PER_BLOCK 4

// ---------------------------------------------------------------------------
// Prepack W2 (f32 [128][128], row = input K, col = output N) into f16 WMMA
// B-fragments: frag[t][k][lane][j], t = N-tile (8), k = K-step (4).
// B layout used here (32x16, contiguous k, A matches): lanes 0-15 hold K=0..15 of column n=lane,
// lanes 16-31 hold K=16..31 of column n=lane-16.
// ---------------------------------------------------------------------------
__global__ void prepack_w2_kernel(const float* __restrict__ W2,
                                  _Float16* __restrict__ frag) {
  int i = blockIdx.x * blockDim.x + threadIdx.x;
  if (i >= 8 * 4 * 32 * 16) return;
  int j    = i & 15;
  int lane = (i >> 4) & 31;
  int k    = (i >> 9) & 3;
  int t    = (i >> 11) & 7;
  int K = k * 32 + ((lane >> 4) << 4) + j;
  int n = t * 16 + (lane & 15);
  frag[i] = (_Float16)W2[K * HDIM + n];
}

// Fast softplus on the TRANS pipe: v_exp_f32 + v_log_f32 (~4 ops instead of a
// ~40-op log1pf libm expansion). exp underflow for very negative x -> log(1)=0,
// the correct limit; large x clamped to identity explicitly.
__device__ __forceinline__ float softplus_f(float x) {
  return (x > 20.f) ? x : __logf(1.f + __expf(x));
}

// Wave-local LDS sync: DS completion + compiler scheduling barrier, without
// coupling the 4 independent waves of the workgroup at a hardware barrier.
__device__ __forceinline__ void wave_sync() {
  __builtin_amdgcn_fence(__ATOMIC_RELEASE, "workgroup");
  __builtin_amdgcn_wave_barrier();
  __builtin_amdgcn_fence(__ATOMIC_ACQUIRE, "workgroup");
}

// ---------------------------------------------------------------------------
// 4 waves per block share one LDS copy of W2; each wave owns a 16-ray group.
// Lane l and l+16 mirror ray (l&15) so EXEC is all-ones across every WMMA.
// ---------------------------------------------------------------------------
__global__ void __launch_bounds__(32 * WAVES_PER_BLOCK) raymarch_mlp_kernel(
    const float* __restrict__ ray_os,
    const float* __restrict__ ray_ds,
    const float* __restrict__ fb,
    const float* __restrict__ W1,
    const float* __restrict__ b1,
    const float* __restrict__ b2,
    const float* __restrict__ W3,
    const float* __restrict__ b3,
    const unsigned char* __restrict__ object_mask,
    const unsigned char* __restrict__ mask_init,
    const _Float16* __restrict__ w2frag,
    float* __restrict__ out, int N)
{
  __shared__ __align__(16) _Float16 sW2[8 * 4 * 32 * 16];          // 32 KB, shared by all waves
  __shared__ __align__(16) _Float16 sH[WAVES_PER_BLOCK][16][HDIM]; // 16 KB, per-wave
  __shared__ float sW1[3 * HDIM];
  __shared__ float sB1[HDIM];
  __shared__ float sB2[HDIM];
  __shared__ float sW3[HDIM];
  __shared__ float sPts[WAVES_PER_BLOCK][16][4];
  __shared__ float sVal[WAVES_PER_BLOCK][16];

  const int tid  = (int)threadIdx.x;
  const int wid  = tid >> 5;          // wave id within block
  const int lane = tid & 31;
  const int r    = lane & 15;
  const int ray  = blockIdx.x * (16 * WAVES_PER_BLOCK) + wid * 16 + r;

  // ---- stage weights into LDS (whole block cooperates, once) ----
  {
    const float4* src = (const float4*)w2frag;   // 16384 halfs = 2048 float4
    float4* dst = (float4*)sW2;
    for (int i = tid; i < 2048; i += 32 * WAVES_PER_BLOCK) dst[i] = src[i];
    for (int i = tid; i < 3 * HDIM; i += 32 * WAVES_PER_BLOCK) sW1[i] = W1[i];
    for (int i = tid; i < HDIM; i += 32 * WAVES_PER_BLOCK) {
      sB1[i] = b1[i];
      sB2[i] = b2[i];
      sW3[i] = W3[i];
    }
  }
  const float b3v = b3[0];
  __syncthreads();           // only block-wide barrier; waves run free after this

  if (ray >= N) return;      // whole-wave uniform for N multiple of 16

  // ---- per-ray state (mirrored in lane and lane+16) ----
  const float ox = ray_os[ray * 3 + 0], oy = ray_os[ray * 3 + 1], oz = ray_os[ray * 3 + 2];
  const float dx = ray_ds[ray * 3 + 0], dy = ray_ds[ray * 3 + 1], dz = ray_ds[ray * 3 + 2];
  float start = fb[ray * 2 + 0];
  float end   = fb[ray * 2 + 1];
  bool uf_s = mask_init[ray] != 0;
  bool uf_e = uf_s;
  const bool obj_m = object_mask[ray] != 0;

  // ---- wave-cooperative SDF eval of the 16 points in sPts[wid] ----
  auto wave_sdf = [&]() {
    // layer 1: 3 -> 128, scalar VALU (2048 values over 32 lanes)
    {
      int m  = lane >> 1;
      int f0 = (lane & 1) << 6;
      float px = sPts[wid][m][0], py = sPts[wid][m][1], pz = sPts[wid][m][2];
      #pragma unroll 4
      for (int f = f0; f < f0 + 64; ++f) {
        float a = fmaf(px, sW1[f],
                  fmaf(py, sW1[HDIM + f],
                  fmaf(pz, sW1[2 * HDIM + f], sB1[f])));
        sH[wid][m][f] = (_Float16)softplus_f(a);
      }
    }
    wave_sync();

    // layer 2: 128x128 via 8 N-tiles x 4 K-steps of v_wmma_f32_16x16x32_f16
    v16h afrag[4];
    {
      int mrow = lane & 15;
      int off  = (lane >> 4) << 4;   // 0 or 16 (A layout half-lane split)
      #pragma unroll
      for (int k = 0; k < 4; ++k) {
        v8h lo = *(const v8h*)&sH[wid][mrow][k * 32 + off];
        v8h hi = *(const v8h*)&sH[wid][mrow][k * 32 + off + 8];
        v16h a;
        #pragma unroll
        for (int j = 0; j < 8; ++j) { a[j] = lo[j]; a[8 + j] = hi[j]; }
        afrag[k] = a;
      }
    }
    v8f acc[8];
    #pragma unroll
    for (int t = 0; t < 8; ++t) {
      v8f c = {};
      #pragma unroll
      for (int k = 0; k < 4; ++k) {
        v16h bfrag = *(const v16h*)&sW2[((t * 4 + k) * 32 + lane) * 16];
        c = __builtin_amdgcn_wmma_f32_16x16x32_f16(
                false, afrag[k], false, bfrag, (short)0, c, false, false);
      }
      acc[t] = c;
    }
    wave_sync();

    // bias + softplus on D accumulators -> sH (layer-3 input)
    // D layout: lane<16 holds n=lane, rows m=0..7; lane>=16 holds n=lane-16, m=8..15
    {
      int mbase = (lane >> 4) << 3;
      #pragma unroll
      for (int t = 0; t < 8; ++t) {
        int f = t * 16 + (lane & 15);
        float bb = sB2[f];
        #pragma unroll
        for (int rr = 0; rr < 8; ++rr)
          sH[wid][mbase + rr][f] = (_Float16)softplus_f(acc[t][rr] + bb);
      }
    }
    wave_sync();

    // layer 3: 128 -> 1 dot + analytic sphere term (vectorized v8h loads)
    if (lane < 16) {
      float s = 0.f;
      #pragma unroll
      for (int c8 = 0; c8 < HDIM / 8; ++c8) {
        v8h hv = *(const v8h*)&sH[wid][lane][c8 * 8];
        const v4f* w4 = (const v4f*)&sW3[c8 * 8];
        #pragma unroll
        for (int j = 0; j < 8; ++j)
          s = fmaf((float)hv[j], sW3[c8 * 8 + j], s);
        (void)w4;
      }
      float qx = sPts[wid][lane][0], qy = sPts[wid][lane][1], qz = sPts[wid][lane][2];
      float nrm = sqrtf(qx * qx + qy * qy + qz * qz);
      sVal[wid][lane] = nrm - 1.f + 0.1f * (s + b3v);
    }
    wave_sync();
  };

  auto eval_at = [&](float t) -> float {
    wave_sync();               // protect sPts/sVal reuse
    if (lane < 16) {
      sPts[wid][lane][0] = ox + t * dx;
      sPts[wid][lane][1] = oy + t * dy;
      sPts[wid][lane][2] = oz + t * dz;
    }
    wave_sync();
    wave_sdf();
    return sVal[wid][r];
  };

  // ---- sphere tracing (matches reference where()-structure) ----
  float v = eval_at(start);
  float next_s = uf_s ? v : 0.f;
  v = eval_at(end);
  float next_e = uf_e ? v : 0.f;

  #pragma unroll 1
  for (int it = 0; it < SPHERE_ITERS; ++it) {
    float curr_s = uf_s ? next_s : 0.f; if (curr_s <= SDF_THR) curr_s = 0.f;
    float curr_e = uf_e ? next_e : 0.f; if (curr_e <= SDF_THR) curr_e = 0.f;
    uf_s = uf_s && (curr_s > SDF_THR);
    uf_e = uf_e && (curr_e > SDF_THR);
    if (uf_s) start += curr_s;
    if (uf_e) end   -= curr_e;
    v = eval_at(start); next_s = uf_s ? v : 0.f;
    v = eval_at(end);   next_e = uf_e ? v : 0.f;
    {  // line search, 1 iter, back = (1-0.5)/2^0 = 0.5
      const float back = 0.5f;
      bool np_s = uf_s && (next_s < 0.f);
      if (np_s) start -= back * curr_s;
      v = eval_at(start); next_s = uf_s ? v : 0.f;
      bool np_e = uf_e && (next_e < 0.f);
      if (np_e) end += back * curr_e;
      v = eval_at(end);   next_e = uf_e ? v : 0.f;
    }
    uf_s = uf_s && (start < end);
    uf_e = uf_e && (start < end);
  }
  uf_s = uf_s && (next_s > SDF_THR);
  const bool work = uf_s;

  // ---- dense sampling: 64 SDF tiles; track argmins + secant seed values ----
  float best_tmp = 3.4e38f; int idx = 0;
  float best_sdf = 3.4e38f; int out_idx = 0;
  float sdf_high = 0.f;      // sdf_val[idx]
  float sdf_low_rec = 0.f;   // sdf_val[idx-1] (valid when idx > 0)
  float prev_sv = 0.f;
  const float span = end - start;
  #pragma unroll 1
  for (int s = 0; s < NSTEPS; ++s) {
    float t  = start + ((float)s * (1.f / 63.f)) * span;
    float sv = eval_at(t);
    float sg  = (sv > 0.f) ? 1.f : ((sv < 0.f) ? -1.f : 0.f);
    float tmp = sg * (float)(NSTEPS - s);
    if (tmp < best_tmp) { best_tmp = tmp; idx = s; sdf_high = sv; sdf_low_rec = prev_sv; }
    if (sv  < best_sdf) { best_sdf = sv; out_idx = s; }
    prev_sv = sv;
  }
  // (idx-1) mod 64 wraps to step 63 when idx == 0; prev_sv now holds sdf[63]
  float sdf_low = (idx == 0) ? prev_sv : sdf_low_rec;

  const bool net_surface = sdf_high < 0.f;
  const bool p_out = !(obj_m && net_surface);
  const int  sel = p_out ? out_idx : idx;
  const float cand_dist = start + ((float)sel * (1.f / 63.f)) * span;
  const bool convergent = work && net_surface;

  // ---- secant refinement ----
  const int idx_low = (idx + NSTEPS - 1) & (NSTEPS - 1);
  float z_high = start + ((float)idx     * (1.f / 63.f)) * span;
  float z_low  = start + ((float)idx_low * (1.f / 63.f)) * span;
  float z_pred = -sdf_low * (z_high - z_low) / (sdf_high - sdf_low + EPS_F) + z_low;
  #pragma unroll 1
  for (int i = 0; i < NSECANT; ++i) {
    float sm = eval_at(z_pred);
    if (sm > 0.f) { z_low  = z_pred; sdf_low  = sm; }
    if (sm < 0.f) { z_high = z_pred; sdf_high = sm; }
    z_pred = -sdf_low * (z_high - z_low) / (sdf_high - sdf_low + EPS_F) + z_low;
  }

  const bool secant_mask = work && net_surface && obj_m;
  float dist = secant_mask ? z_pred : (work ? cand_dist : 0.f);
  float fx, fy, fz;
  if (secant_mask)      { fx = ox + z_pred    * dx; fy = oy + z_pred    * dy; fz = oz + z_pred    * dz; }
  else if (work)        { fx = ox + cand_dist * dx; fy = oy + cand_dist * dy; fz = oz + cand_dist * dz; }
  else                  { fx = 0.f; fy = 0.f; fz = 0.f; }

  if (lane < 16) {
    out[ray * 5 + 0] = fx;
    out[ray * 5 + 1] = fy;
    out[ray * 5 + 2] = fz;
    out[ray * 5 + 3] = dist;
    out[ray * 5 + 4] = convergent ? 1.f : 0.f;
  }
}

// ---------------------------------------------------------------------------
extern "C" void kernel_launch(void* const* d_in, const int* in_sizes, int n_in,
                              void* d_out, int out_size, void* d_ws, size_t ws_size,
                              hipStream_t stream) {
  (void)n_in; (void)out_size; (void)ws_size;
  const float* ray_os = (const float*)d_in[0];
  const float* ray_ds = (const float*)d_in[1];
  const float* fb     = (const float*)d_in[2];
  const float* W1     = (const float*)d_in[3];
  const float* b1     = (const float*)d_in[4];
  const float* W2     = (const float*)d_in[5];
  const float* b2     = (const float*)d_in[6];
  const float* W3     = (const float*)d_in[7];
  const float* b3     = (const float*)d_in[8];
  const unsigned char* obj = (const unsigned char*)d_in[9];
  const unsigned char* msk = (const unsigned char*)d_in[10];

  const int N = in_sizes[0] / 3;                 // 16384
  _Float16* w2frag = (_Float16*)d_ws;            // 32 KB of prepacked fragments

  prepack_w2_kernel<<<(8 * 4 * 32 * 16 + 255) / 256, 256, 0, stream>>>(W2, w2frag);

  const int rays_per_block = 16 * WAVES_PER_BLOCK;
  const int nblocks = (N + rays_per_block - 1) / rays_per_block;
  raymarch_mlp_kernel<<<nblocks, 32 * WAVES_PER_BLOCK, 0, stream>>>(
      ray_os, ray_ds, fb, W1, b1, b2, W3, b3, obj, msk, w2frag,
      (float*)d_out, N);
}
